// HGRN2Block_7421703488078
// MI455X (gfx1250) — matrix operand order repaired and checked by reading, not compiled, not run on hardware
//
#include <hip/hip_runtime.h>
#include <hip/hip_bf16.h>

// ---------------------------------------------------------------------------
// HGRN2Block live path on MI455X (gfx1250):
//   G = x @ Wp[:,0:1024] + b[0:1024]
//   O = x @ Wp[:,2048:3072] + b[2048:3072]
//   H = G * sigmoid(O)              (bf16, in d_ws)
//   out = H @ W_out + b_out         (fp32, d_out)
// bf16 WMMA (V_WMMA_F32_16X16X32_BF16) with fp32 accumulate.
// Double-buffered LDS, one barrier per K-step, packed b32 transpose stores.
// ---------------------------------------------------------------------------

typedef __attribute__((ext_vector_type(16))) __bf16 v16bf;
typedef __attribute__((ext_vector_type(8)))  __bf16 v8bf;
typedef __attribute__((ext_vector_type(4)))  __bf16 v4bf;
typedef __attribute__((ext_vector_type(2)))  __bf16 v2bf;
typedef __attribute__((ext_vector_type(8)))  float  v8f;

#define D_MODEL 1024
#define WPROJ_LD 3072           // W_proj row stride (3*D)
#define NKSTEP (D_MODEL / 32)   // 32 K-steps

static __device__ __forceinline__ v16bf cat8(v8bf lo, v8bf hi) {
  return __builtin_shufflevector(lo, hi, 0,1,2,3,4,5,6,7,8,9,10,11,12,13,14,15);
}

// ---------------------------------------------------------------------------
// Kernel 1: fused projection + gate.  Block tile 128(M) x 64(N), K-step 32.
// 256 threads = 8 waves; wave grid 2(M) x 4(N); each wave: 64x16 strip with
// dual accumulators (gate & output-gate) sharing the A fragments.
// ---------------------------------------------------------------------------
__global__ __launch_bounds__(256)
void hgrn2_proj_gate(const float* __restrict__ x,
                     const float* __restrict__ Wp,
                     const float* __restrict__ bp,
                     __bf16* __restrict__ H)
{
  __shared__ __align__(16) __bf16 ldsA [2][128 * 32];  // [m][k]
  __shared__ __align__(16) __bf16 ldsBg[2][ 64 * 32];  // [n][k] transposed
  __shared__ __align__(16) __bf16 ldsBo[2][ 64 * 32];  // [n][k] transposed

  const int t     = threadIdx.x;
  const int lane  = t & 31;
  const int wave  = t >> 5;
  const int waveM = wave >> 2;       // 0..1 -> M offset 0/64
  const int waveN = wave & 3;        // 0..3 -> N offset *16
  const int khalf = lane >> 4;
  const int l16   = lane & 15;
  const int mbase = blockIdx.x * 128;
  const int nbase = blockIdx.y * 64;

  auto fill = [&](int buf, int kbase) {
    // A tile: 128x32 fp32 -> bf16 (4 float4 per thread, b64 LDS stores)
    #pragma unroll
    for (int j = 0; j < 4; ++j) {
      const int f   = t + j * 256;
      const int row = f >> 3;
      const int c4  = (f & 7) << 2;
      float4 v = *(const float4*)(x + (size_t)(mbase + row) * D_MODEL + kbase + c4);
      v4bf p; p.x = (__bf16)v.x; p.y = (__bf16)v.y; p.z = (__bf16)v.z; p.w = (__bf16)v.w;
      *(v4bf*)(&ldsA[buf][row * 32 + c4]) = p;
    }
    // Bg/Bo tiles: 32x64 fp32 -> bf16, transposed; each thread packs a pair
    // of K rows per n -> b32 LDS stores (half the DS-store count of b16).
    {
      const int kp = t >> 4;           // 0..15 -> k rows 2kp, 2kp+1
      const int n4 = (t & 15) << 2;    // 0..60
      const float* w0 = Wp + (size_t)(kbase + 2 * kp) * WPROJ_LD + nbase + n4;
      float4 g0 = *(const float4*)(w0);
      float4 g1 = *(const float4*)(w0 + WPROJ_LD);
      float4 o0 = *(const float4*)(w0 + 2 * D_MODEL);
      float4 o1 = *(const float4*)(w0 + WPROJ_LD + 2 * D_MODEL);
      __bf16* bg = &ldsBg[buf][0];
      __bf16* bo = &ldsBo[buf][0];
      const int kk = 2 * kp;
      v2bf p;
      p.x = (__bf16)g0.x; p.y = (__bf16)g1.x; *(v2bf*)(bg + (n4+0)*32 + kk) = p;
      p.x = (__bf16)g0.y; p.y = (__bf16)g1.y; *(v2bf*)(bg + (n4+1)*32 + kk) = p;
      p.x = (__bf16)g0.z; p.y = (__bf16)g1.z; *(v2bf*)(bg + (n4+2)*32 + kk) = p;
      p.x = (__bf16)g0.w; p.y = (__bf16)g1.w; *(v2bf*)(bg + (n4+3)*32 + kk) = p;
      p.x = (__bf16)o0.x; p.y = (__bf16)o1.x; *(v2bf*)(bo + (n4+0)*32 + kk) = p;
      p.x = (__bf16)o0.y; p.y = (__bf16)o1.y; *(v2bf*)(bo + (n4+1)*32 + kk) = p;
      p.x = (__bf16)o0.z; p.y = (__bf16)o1.z; *(v2bf*)(bo + (n4+2)*32 + kk) = p;
      p.x = (__bf16)o0.w; p.y = (__bf16)o1.w; *(v2bf*)(bo + (n4+3)*32 + kk) = p;
    }
  };

  v8f accG[4], accO[4];
  #pragma unroll
  for (int i = 0; i < 4; ++i) { accG[i] = (v8f){}; accO[i] = (v8f){}; }

  fill(0, 0);

  for (int kt = 0; kt < NKSTEP; ++kt) {
    const int cur = kt & 1;
    __syncthreads();

    // ---- issue fragment loads from current buffer first ----
    const __bf16* A  = &ldsA [cur][0];
    const __bf16* Bg = &ldsBg[cur][0];
    const __bf16* Bo = &ldsBo[cur][0];
    const int nrow = waveN * 16 + l16;
    const __bf16* bgp = Bg + nrow * 32 + khalf * 16;  // B: 16 contiguous K/lane
    const __bf16* bop = Bo + nrow * 32 + khalf * 16;
    v16bf bg = cat8(*(const v8bf*)(bgp), *(const v8bf*)(bgp + 8));
    v16bf bo = cat8(*(const v8bf*)(bop), *(const v8bf*)(bop + 8));
    v16bf afrag[4];
    #pragma unroll
    for (int mi = 0; mi < 4; ++mi) {
      const int mrow = waveM * 64 + mi * 16 + l16;
      const __bf16* ap = A + mrow * 32 + khalf * 16;  // A: 16 contiguous K/lane
      afrag[mi] = cat8(*(const v8bf*)(ap), *(const v8bf*)(ap + 8));
    }

    // ---- overlap: prefetch next tile into the other buffer ----
    if (kt + 1 < NKSTEP) fill(cur ^ 1, (kt + 1) * 32);

    // ---- 8 WMMAs per K-step per wave ----
    #pragma unroll
    for (int mi = 0; mi < 4; ++mi) {
      accG[mi] = __builtin_amdgcn_wmma_f32_16x16x32_bf16(
          false, afrag[mi], false, bg, (short)0, accG[mi], false, false);
      accO[mi] = __builtin_amdgcn_wmma_f32_16x16x32_bf16(
          false, afrag[mi], false, bo, (short)0, accO[mi], false, false);
    }
  }

  // ---- epilogue: bias + sigmoid gate, write bf16 H ----
  const int ng   = nbase + waveN * 16 + l16;
  const float bgb = bp[ng];
  const float bob = bp[2 * D_MODEL + ng];
  #pragma unroll
  for (int mi = 0; mi < 4; ++mi) {
    #pragma unroll
    for (int r = 0; r < 8; ++r) {
      const int mg = mbase + waveM * 64 + mi * 16 + khalf * 8 + r;
      const float g = accG[mi][r] + bgb;
      const float o = accO[mi][r] + bob;
      const float h = g * __frcp_rn(1.0f + __expf(-o));
      H[(size_t)mg * D_MODEL + ng] = (__bf16)h;
    }
  }
}

// ---------------------------------------------------------------------------
// Kernel 2: out = H @ W_out + b_out.  Block tile 128x128, wave tile 64x32.
// ---------------------------------------------------------------------------
__global__ __launch_bounds__(256)
void hgrn2_out_proj(const __bf16* __restrict__ H,
                    const float* __restrict__ Wout,
                    const float* __restrict__ bout,
                    float* __restrict__ out)
{
  __shared__ __align__(16) __bf16 ldsA[2][128 * 32];  // [m][k]
  __shared__ __align__(16) __bf16 ldsB[2][128 * 32];  // [n][k] transposed

  const int t     = threadIdx.x;
  const int lane  = t & 31;
  const int wave  = t >> 5;
  const int waveM = wave >> 2;       // 0..1
  const int waveN = wave & 3;        // 0..3 -> N offset *32
  const int khalf = lane >> 4;
  const int l16   = lane & 15;
  const int mbase = blockIdx.x * 128;
  const int nbase = blockIdx.y * 128;

  auto fill = [&](int buf, int kbase) {
    // A tile: 128x32 bf16 straight copy (16B chunks, 2 per thread)
    #pragma unroll
    for (int j = 0; j < 2; ++j) {
      const int f   = t + j * 256;
      const int row = f >> 2;
      const int c8  = (f & 3) << 3;
      *(v8bf*)(&ldsA[buf][row * 32 + c8]) =
          *(const v8bf*)(H + (size_t)(mbase + row) * D_MODEL + kbase + c8);
    }
    // B tile: 32x128 fp32 -> bf16 transposed, packed K-pair b32 stores
    #pragma unroll
    for (int j = 0; j < 2; ++j) {
      const int f  = t + j * 256;      // 0..511
      const int kp = f >> 5;           // 0..15 -> k rows 2kp, 2kp+1
      const int n4 = (f & 31) << 2;    // 0..124
      const float* w0 = Wout + (size_t)(kbase + 2 * kp) * D_MODEL + nbase + n4;
      float4 v0 = *(const float4*)(w0);
      float4 v1 = *(const float4*)(w0 + D_MODEL);
      __bf16* bb = &ldsB[buf][0];
      const int kk = 2 * kp;
      v2bf p;
      p.x = (__bf16)v0.x; p.y = (__bf16)v1.x; *(v2bf*)(bb + (n4+0)*32 + kk) = p;
      p.x = (__bf16)v0.y; p.y = (__bf16)v1.y; *(v2bf*)(bb + (n4+1)*32 + kk) = p;
      p.x = (__bf16)v0.z; p.y = (__bf16)v1.z; *(v2bf*)(bb + (n4+2)*32 + kk) = p;
      p.x = (__bf16)v0.w; p.y = (__bf16)v1.w; *(v2bf*)(bb + (n4+3)*32 + kk) = p;
    }
  };

  v8f acc[4][2];
  #pragma unroll
  for (int mi = 0; mi < 4; ++mi)
    #pragma unroll
    for (int ni = 0; ni < 2; ++ni) acc[mi][ni] = (v8f){};

  fill(0, 0);

  for (int kt = 0; kt < NKSTEP; ++kt) {
    const int cur = kt & 1;
    __syncthreads();

    const __bf16* A = &ldsA[cur][0];
    const __bf16* B = &ldsB[cur][0];
    v16bf bfrag[2];
    #pragma unroll
    for (int ni = 0; ni < 2; ++ni) {
      const int nrow = waveN * 32 + ni * 16 + l16;
      const __bf16* bptr = B + nrow * 32 + khalf * 16;
      bfrag[ni] = cat8(*(const v8bf*)(bptr), *(const v8bf*)(bptr + 8));
    }
    v16bf afrag[4];
    #pragma unroll
    for (int mi = 0; mi < 4; ++mi) {
      const int mrow = waveM * 64 + mi * 16 + l16;
      const __bf16* ap = A + mrow * 32 + khalf * 16;
      afrag[mi] = cat8(*(const v8bf*)(ap), *(const v8bf*)(ap + 8));
    }

    if (kt + 1 < NKSTEP) fill(cur ^ 1, (kt + 1) * 32);

    #pragma unroll
    for (int mi = 0; mi < 4; ++mi)
      #pragma unroll
      for (int ni = 0; ni < 2; ++ni)
        acc[mi][ni] = __builtin_amdgcn_wmma_f32_16x16x32_bf16(
            false, afrag[mi], false, bfrag[ni], (short)0, acc[mi][ni], false, false);
  }

  // ---- epilogue: bias, fp32 store ----
  #pragma unroll
  for (int ni = 0; ni < 2; ++ni) {
    const int ng = nbase + waveN * 32 + ni * 16 + l16;
    const float bb = bout[ng];
    #pragma unroll
    for (int mi = 0; mi < 4; ++mi) {
      #pragma unroll
      for (int r = 0; r < 8; ++r) {
        const int mg = mbase + waveM * 64 + mi * 16 + khalf * 8 + r;
        out[(size_t)mg * D_MODEL + ng] = acc[mi][ni][r] + bb;
      }
    }
  }
}

// ---------------------------------------------------------------------------
extern "C" void kernel_launch(void* const* d_in, const int* in_sizes, int n_in,
                              void* d_out, int out_size, void* d_ws, size_t ws_size,
                              hipStream_t stream) {
  const float* x    = (const float*)d_in[0];   // [16384, 1024]
  const float* Wp   = (const float*)d_in[1];   // [1024, 3072]
  const float* bp   = (const float*)d_in[2];   // [3072]
  const float* Wout = (const float*)d_in[3];   // [1024, 1024]
  const float* bout = (const float*)d_in[4];   // [1024]
  float* out = (float*)d_out;

  const int M = in_sizes[0] / D_MODEL;         // 16384 rows
  __bf16* H = (__bf16*)d_ws;                   // M*1024 bf16 = 32 MB scratch

  dim3 blk(256);
  dim3 grid1(M / 128, D_MODEL / 64);           // fused proj+gate
  hgrn2_proj_gate<<<grid1, blk, 0, stream>>>(x, Wp, bp, H);

  dim3 grid2(M / 128, D_MODEL / 128);          // output projection
  hgrn2_out_proj<<<grid2, blk, 0, stream>>>(H, Wout, bout, out);
}
